// MultiHeadAttention_76510547410991
// MI455X (gfx1250) — matrix operand order repaired and checked by reading, not compiled, not run on hardware
//
#include <hip/hip_runtime.h>
#include <stdint.h>

// Problem sizes (from reference)
#define SEQ   32768
#define FDIM  256
#define FH    2048        // F*H
#define ROWS  128         // S-rows per workgroup
#define WAVES 8           // 256 threads, wave32

typedef __attribute__((ext_vector_type(16))) __bf16 v16bf;
typedef __attribute__((ext_vector_type(8)))  __bf16 v8bf;
typedef __attribute__((ext_vector_type(4)))  __bf16 v4bf;
typedef __attribute__((ext_vector_type(8)))  float  v8f;
typedef __attribute__((ext_vector_type(4)))  float  v4f;

__device__ __forceinline__ v16bf cat16(v8bf lo, v8bf hi) {
  return __builtin_shufflevector(lo, hi, 0,1,2,3,4,5,6,7,8,9,10,11,12,13,14,15);
}

// A-operand (16x32 bf16) from a 16-row strip in LDS, row stride `ld` elements.
// Lanes 0-15: row=lane,    K = {k0..k0+7, k0+16..k0+23}
// Lanes 16-31: row=lane-16, K = {k0+8..k0+15, k0+24..k0+31}
__device__ __forceinline__ v16bf load_a_lds(const __bf16* strip, int lane, int ld, int k0) {
  const __bf16* r = strip + (lane & 15) * ld;
  const int off = k0 + ((lane < 16) ? 0 : 8);
  v8bf lo = *(const v8bf*)(r + off);
  v8bf hi = *(const v8bf*)(r + off + 16);
  return cat16(lo, hi);
}

// B-operand (32x16 bf16) from a K-major bf16 weight matrix W[ldw wide] in global:
// column = j+lane%16, element s holds K = k0 + 8*(lane/16) + (s<8 ? s : s+8).
__device__ __forceinline__ v16bf load_b_glb(const __bf16* __restrict__ W, int lane, int ldw,
                                            int k0, int j) {
  v16bf b;
  const __bf16* p = W + (size_t)k0 * ldw + j + (lane & 15);
  const int kb = (lane >> 4) * 8;
  #pragma unroll
  for (int s = 0; s < 16; ++s) b[s] = p[(size_t)(kb + (s < 8 ? s : s + 8)) * ldw];
  return b;
}

// B-operand from a staged 32x256 K-major chunk in LDS.
__device__ __forceinline__ v16bf load_b_lds(const __bf16* Cbuf, int lane, int t) {
  v16bf b;
  const __bf16* p = Cbuf + t * 16 + (lane & 15);
  const int kb = (lane >> 4) * 8;
  #pragma unroll
  for (int s = 0; s < 16; ++s) b[s] = p[(kb + (s < 8 ? s : s + 8)) * FDIM];
  return b;
}

// ---- CDNA5 async global->LDS copy (tracked by ASYNCcnt) ----
__device__ __forceinline__ void async_ld_b128(const void* g, const void* lds_ptr) {
  uint32_t loff = (uint32_t)(uintptr_t)lds_ptr;        // low 32 bits of generic ptr = LDS offset
  uint64_t ga   = (uint64_t)(uintptr_t)g;
  asm volatile("global_load_async_to_lds_b128 %0, %1, off"
               :: "v"(loff), "v"(ga) : "memory");
}

__device__ __forceinline__ void wait_async0() {
#if __has_builtin(__builtin_amdgcn_s_wait_asynccnt)
  __builtin_amdgcn_s_wait_asynccnt(0);
#else
  asm volatile("s_wait_asynccnt 0x0" ::: "memory");
#endif
}

// Cooperatively stage WcT[jc:jc+32, 0:256] (16 KB) into an LDS buffer: 4 async b128/thread.
__device__ __forceinline__ void stage_wc(const __bf16* __restrict__ WcT, __bf16* Cbuf,
                                         int jc, int tid) {
  #pragma unroll
  for (int i = 0; i < 4; ++i) {
    int chunk = tid + i * 256;                         // 1024 b128 chunks total
    async_ld_b128(WcT + (size_t)jc * FDIM + (size_t)chunk * 8, Cbuf + chunk * 8);
  }
}

// ---- prep: transpose + fp32->bf16 convert the three weight matrices ----
__global__ void prep_weights(const float* __restrict__ Wq, const float* __restrict__ Wv,
                             const float* __restrict__ Wc,
                             unsigned short* __restrict__ wqt_u, unsigned short* __restrict__ wvt_u,
                             unsigned short* __restrict__ wct_u) {
  __bf16* WqT = (__bf16*)wqt_u;   // [256][2048]  (K-major for q/v projections)
  __bf16* WvT = (__bf16*)wvt_u;   // [256][2048]
  __bf16* WcT = (__bf16*)wct_u;   // [2048][256]  (K-major for output projection)
  int i = blockIdx.x * 256 + threadIdx.x;
  if (i < FDIM * FH) {
    int k = i / FH, j = i % FH;          // WqT[k][j] = Wq[j][k]
    WqT[i] = (__bf16)Wq[(size_t)j * FDIM + k];
    WvT[i] = (__bf16)Wv[(size_t)j * FDIM + k];
    int c = i / FDIM, n = i % FDIM;      // WcT[c][n] = Wc[n][c]
    WcT[i] = (__bf16)Wc[(size_t)n * FH + c];
  }
}

// ---- fused: ((X Wq^T + bq) .* (V Wv^T + bv)) Wc^T + bc ----
__global__ __launch_bounds__(256, 1)
void fused_qv_proj(const float* __restrict__ X, const float* __restrict__ V,
                   const unsigned short* __restrict__ wqt_u,
                   const unsigned short* __restrict__ wvt_u,
                   const unsigned short* __restrict__ wct_u,
                   const float* __restrict__ bq, const float* __restrict__ bv,
                   const float* __restrict__ bc, float* __restrict__ out) {
  const __bf16* WqT = (const __bf16*)wqt_u;
  const __bf16* WvT = (const __bf16*)wvt_u;
  const __bf16* WcT = (const __bf16*)wct_u;

  __shared__ __bf16 Xs[ROWS * FDIM];        // 64 KB: X tile, bf16
  __shared__ __bf16 Vs[ROWS * FDIM];        // 64 KB: V tile, bf16
  __shared__ __bf16 Ps[WAVES][16 * 32];     // 8 KB: per-wave C->A relayout scratch
  __shared__ __bf16 Cs0[32 * FDIM];         // 16 KB: WcT chunk, double buffer 0
  __shared__ __bf16 Cs1[32 * FDIM];         // 16 KB: WcT chunk, double buffer 1

  const int tid  = threadIdx.x;
  const int lane = tid & 31;
  const int w    = tid >> 5;                // wave id 0..7
  const int col  = lane & 15;
  const size_t row0 = (size_t)blockIdx.x * ROWS;

  // ---- kick off async staging of the first WcT chunk ----
  stage_wc(WcT, Cs0, 0, tid);

  // ---- stage X/V tile into LDS as bf16 (float4 loads, 8B LDS stores) ----
  #pragma unroll
  for (int i = 0; i < (ROWS * FDIM) / (4 * 256); ++i) {   // 32 iters
    int idx = tid + i * 256;                               // float4 index
    v4f x4 = *(const v4f*)(X + row0 * FDIM + (size_t)idx * 4);
    v4f v4 = *(const v4f*)(V + row0 * FDIM + (size_t)idx * 4);
    v4bf xb, vb;
    #pragma unroll
    for (int c = 0; c < 4; ++c) { xb[c] = (__bf16)x4[c]; vb[c] = (__bf16)v4[c]; }
    *(v4bf*)(Xs + idx * 4) = xb;
    *(v4bf*)(Vs + idx * 4) = vb;
  }
  wait_async0();
  __syncthreads();

  const __bf16* xstrip = Xs + w * 16 * FDIM;
  const __bf16* vstrip = Vs + w * 16 * FDIM;
  __bf16* ps = &Ps[w][0];

  // ---- pull this wave's A-operand strips into REGISTERS once (immune to in-loop
  //      barriers; 128 VGPRs) so the jc loop never re-reads them from LDS ----
  v16bf axr[8], avr[8];
  #pragma unroll
  for (int kk = 0; kk < 8; ++kk) {
    axr[kk] = load_a_lds(xstrip, lane, FDIM, kk * 32);
    avr[kk] = load_a_lds(vstrip, lane, FDIM, kk * 32);
  }

  // fp32 accumulator: full 16x256 output strip (16 C-tiles of 8 VGPRs)
  v8f acc[16];
  #pragma unroll
  for (int t = 0; t < 16; ++t) {
    float b = bc[t * 16 + col];
    #pragma unroll
    for (int r = 0; r < 8; ++r) acc[t][r] = b;
  }

  const int prow = (lane >> 4) * 8;   // C-layout: M = r + 8*(lane>=16)

  for (int jc = 0; jc < FH; jc += 32) {
    const __bf16* Crd = (jc & 32) ? Cs1 : Cs0;
    __bf16*       Cwr = (jc & 32) ? Cs0 : Cs1;

    // async-stage next chunk's WcT into the other buffer (overlaps with WMMAs below)
    if (jc + 32 < FH) stage_wc(WcT, Cwr, jc + 32, tid);

    // q/v projection for columns [jc, jc+32), 16 at a time (B from global/L0 path)
    #pragma unroll
    for (int half = 0; half < 2; ++half) {
      const int j = jc + half * 16;
      v8f q = {}, vv = {};
      #pragma unroll
      for (int kk = 0; kk < 8; ++kk) {
        v16bf bqop = load_b_glb(WqT, lane, FH, kk * 32, j);
        v16bf bvop = load_b_glb(WvT, lane, FH, kk * 32, j);
        q  = __builtin_amdgcn_wmma_f32_16x16x32_bf16(false, axr[kk], false, bqop, (short)0, q,  false, false);
        vv = __builtin_amdgcn_wmma_f32_16x16x32_bf16(false, avr[kk], false, bvop, (short)0, vv, false, false);
      }
      const float bqs = bq[j + col];
      const float bvs = bv[j + col];
      // p = (q+bq)*(v+bv); re-layout C->A through per-wave LDS scratch (same-wave DS is in-order)
      #pragma unroll
      for (int r = 0; r < 8; ++r) {
        float p = (q[r] + bqs) * (vv[r] + bvs);
        ps[(prow + r) * 32 + half * 16 + col] = (__bf16)p;
      }
    }

    // output projection: acc[16x256] += P[16x32] * WcT[jc:jc+32, 0:256] (B from LDS path)
    v16bf ap = load_a_lds(ps, lane, 32, 0);
    #pragma unroll
    for (int t = 0; t < 16; ++t) {
      v16bf bcop = load_b_lds(Crd, lane, t);
      acc[t] = __builtin_amdgcn_wmma_f32_16x16x32_bf16(false, ap, false, bcop, (short)0, acc[t], false, false);
    }

    // next chunk's staging must be complete (and visible block-wide) before it is read
    wait_async0();
    __syncthreads();
  }

  // ---- write out: C layout -> row = w*16 + r + 8*(lane>=16), col = t*16 + lane%16
  const size_t rbase = row0 + w * 16 + prow;
  #pragma unroll
  for (int t = 0; t < 16; ++t)
    #pragma unroll
    for (int r = 0; r < 8; ++r)
      out[(rbase + r) * FDIM + t * 16 + col] = acc[t][r];
}

extern "C" void kernel_launch(void* const* d_in, const int* in_sizes, int n_in,
                              void* d_out, int out_size, void* d_ws, size_t ws_size,
                              hipStream_t stream) {
  const float* X  = (const float*)d_in[0];   // query_key_input [S,F]
  const float* V  = (const float*)d_in[1];   // value           [S,F]
  const float* Wq = (const float*)d_in[2];   // [FH,F]
  const float* bq = (const float*)d_in[3];   // [FH]
  // d_in[4]=Wk, d_in[5]=bk are dead in the reference math
  const float* Wv = (const float*)d_in[6];   // [FH,F]
  const float* bv = (const float*)d_in[7];   // [FH]
  const float* Wc = (const float*)d_in[8];   // [F,FH]
  const float* bc = (const float*)d_in[9];   // [F]
  float* out = (float*)d_out;                // [S,F]

  // workspace: three bf16 transposed weight matrices, 1 MB each
  unsigned short* WqT = (unsigned short*)d_ws;
  unsigned short* WvT = WqT + (size_t)FDIM * FH;
  unsigned short* WcT = WvT + (size_t)FDIM * FH;

  prep_weights<<<(FDIM * FH + 255) / 256, 256, 0, stream>>>(Wq, Wv, Wc, WqT, WvT, WcT);
  fused_qv_proj<<<SEQ / ROWS, 256, 0, stream>>>(X, V, WqT, WvT, WcT, bq, bv, bc, out);
}
